// GroupedQueryAttention_47519518163008
// MI455X (gfx1250) — matrix operand order repaired and checked by reading, not compiled, not run on hardware
//
#include <hip/hip_runtime.h>
#include <hip/hip_bf16.h>
#include <math.h>

// ---------------------------------------------------------------------------
// GQA attention for MI455X (gfx1250, wave32, WMMA 16x16x32 bf16)
// B=2 S=2048 D=2048 H=32 KVH=8 HD=64 G=4
// ---------------------------------------------------------------------------

#define B_   2
#define S_   2048
#define D_   2048
#define H_   32
#define KVH_ 8
#define HD_  64

typedef __attribute__((ext_vector_type(16))) __bf16 v16bf;
typedef __attribute__((ext_vector_type(8)))  float  v8f;
typedef __attribute__((ext_vector_type(4)))  int    v4i;

// ---- CDNA5 async global->LDS path (guarded; sync fallback keeps compiles) --
#if defined(__has_builtin)
#if __has_builtin(__builtin_amdgcn_global_load_async_to_lds_b128)
#define ASYNC_LDS 1
#endif
#endif
#ifndef ASYNC_LDS
#define ASYNC_LDS 0
#endif

#if ASYNC_LDS
typedef __attribute__((address_space(1))) v4i* as1_v4i_p;
typedef __attribute__((address_space(3))) v4i* as3_v4i_p;

static __device__ __forceinline__ void async_copy_b128(__bf16* lds_dst,
                                                       const __bf16* g_src) {
  // Signature (from hipcc diagnostic): (v4i as1*, v4i as3*, imm offset, imm cpol)
  __builtin_amdgcn_global_load_async_to_lds_b128(
      (as1_v4i_p)(g_src), (as3_v4i_p)(lds_dst), 0, 0);
}
#endif

static __device__ __forceinline__ void wait_async0() {
#if ASYNC_LDS
#if __has_builtin(__builtin_amdgcn_s_wait_asynccnt)
  __builtin_amdgcn_s_wait_asynccnt(0);
#else
  asm volatile("s_wait_asynccnt 0x0" ::: "memory");
#endif
#endif
}

// Load a 16-element bf16 fragment from two 16-byte segments (A- or B-matrix
// fragment; callers here use contiguous k = 16*half + s). Works for global or LDS pointers.
static __device__ __forceinline__ v16bf load_frag(const __bf16* p0, const __bf16* p1) {
  union { uint4 u[2]; v16bf v; } t;
  t.u[0] = *reinterpret_cast<const uint4*>(p0);
  t.u[1] = *reinterpret_cast<const uint4*>(p1);
  return t.v;
}

static __device__ __forceinline__ v8f wmma_bf16(v16bf a, v16bf b, v8f c) {
  return __builtin_amdgcn_wmma_f32_16x16x32_bf16(false, a, false, b, (short)0, c,
                                                 false, false);
}

// ---------------------------------------------------------------------------
// fp32 -> bf16 elementwise convert
// ---------------------------------------------------------------------------
__global__ __launch_bounds__(256) void cvt_f32_bf16(const float* __restrict__ src,
                                                    __bf16* __restrict__ dst, int n) {
  int i = blockIdx.x * blockDim.x + threadIdx.x;
  if (i < n) dst[i] = (__bf16)src[i];
}

// ---------------------------------------------------------------------------
// Transpose fp32 (K x N) -> bf16 (N x K).  Writes are fully coalesced.
// ---------------------------------------------------------------------------
__global__ __launch_bounds__(256) void transpose_bf16(const float* __restrict__ src,
                                                      __bf16* __restrict__ dst,
                                                      int K, int N) {
  int idx = blockIdx.x * blockDim.x + threadIdx.x;  // idx over N*K
  int k  = idx % K;
  int nn = idx / K;
  if (nn < N) dst[(size_t)nn * K + k] = (__bf16)src[(size_t)k * N + nn];
}

// ---------------------------------------------------------------------------
// LDS-staged double-buffered WMMA GEMM:
//   C(MxN fp32, row-major) = A(MxK bf16, row-major) * B,  B given as BT (NxK).
// Block = 256 threads = 8 waves. Block tile 128(M) x 64(N) x 64(K).
// A tile (128x64) + B tile (64x64) double-buffered in LDS (~55KB of the WGP's
// 320KB). Stage s+1 is issued (async-to-LDS) before computing stage s, so
// global latency hides behind the WMMA chain. All 10 fragments of a stage are
// preloaded into registers (20x ds_load_b128) before issuing 8 back-to-back
// v_wmma, so the MMA pipe is not stalled by per-WMMA dscnt waits.
// ---------------------------------------------------------------------------
#define BM 128
#define BN 64
#define BK 64
#define LDA 72  // padded row pitch (elements) to spread LDS banks
#define LDB 72

__global__ __launch_bounds__(256) void gemm_bf16_wmma(const __bf16* __restrict__ A,
                                                      const __bf16* __restrict__ BT,
                                                      float* __restrict__ C,
                                                      int M, int N, int K) {
  __shared__ __align__(16) __bf16 As[2][BM * LDA];
  __shared__ __align__(16) __bf16 Bs[2][BN * LDB];

  const int tid  = threadIdx.x;
  const int lane = tid & 31;
  const int wave = tid >> 5;
  const int m_blk = blockIdx.y * BM;
  const int n_blk = blockIdx.x * BN;
  const int col  = lane & 15;
  const int half = lane >> 4;

  // staging work split: A row = tid/2 (64B each), B row = tid/4 (32B each)
  const int arow = tid >> 1, aoff = (tid & 1) * 32;
  const int brow = tid >> 2, boff = (tid & 3) * 16;

  v8f acc[4] = {};

  auto stage = [&](int kk, int buf) {
    const __bf16* ga = A + (size_t)(m_blk + arow) * K + kk + aoff;
    __bf16*       la = &As[buf][arow * LDA + aoff];
    const __bf16* gb = BT + (size_t)(n_blk + brow) * K + kk + boff;
    __bf16*       lb = &Bs[buf][brow * LDB + boff];
#if ASYNC_LDS
    async_copy_b128(la,      ga);
    async_copy_b128(la + 8,  ga + 8);
    async_copy_b128(la + 16, ga + 16);
    async_copy_b128(la + 24, ga + 24);
    async_copy_b128(lb,      gb);
    async_copy_b128(lb + 8,  gb + 8);
#else
    uint4 a0 = *(const uint4*)(ga);
    uint4 a1 = *(const uint4*)(ga + 8);
    uint4 a2 = *(const uint4*)(ga + 16);
    uint4 a3 = *(const uint4*)(ga + 24);
    uint4 b0 = *(const uint4*)(gb);
    uint4 b1 = *(const uint4*)(gb + 8);
    *(uint4*)(la)      = a0;
    *(uint4*)(la + 8)  = a1;
    *(uint4*)(la + 16) = a2;
    *(uint4*)(la + 24) = a3;
    *(uint4*)(lb)      = b0;
    *(uint4*)(lb + 8)  = b1;
#endif
  };

  // Preload all fragments of a 64-K stage, then 8 back-to-back WMMAs.
  auto compute = [&](int buf) {
    const __bf16* abase = &As[buf][(wave * 16 + col) * LDA];
    v16bf af0 = load_frag(abase + half * 16,      abase + half * 16 + 8);
    v16bf af1 = load_frag(abase + 32 + half * 16, abase + 32 + half * 16 + 8);
    v16bf bf[8];
#pragma unroll
    for (int t = 0; t < 4; ++t) {
      const __bf16* bbase = &Bs[buf][(t * 16 + col) * LDB];
      bf[t]     = load_frag(bbase + half * 16,      bbase + half * 16 + 8);
      bf[t + 4] = load_frag(bbase + 32 + half * 16, bbase + 32 + half * 16 + 8);
    }
#pragma unroll
    for (int t = 0; t < 4; ++t) acc[t] = wmma_bf16(af0, bf[t], acc[t]);
#pragma unroll
    for (int t = 0; t < 4; ++t) acc[t] = wmma_bf16(af1, bf[t + 4], acc[t]);
  };

  stage(0, 0);
  wait_async0();
  __syncthreads();

  const int nst = K / BK;
  for (int s = 0; s < nst; ++s) {
    const int cur = s & 1;
    if (s + 1 < nst) stage((s + 1) * BK, cur ^ 1);
    compute(cur);
    wait_async0();
    __syncthreads();
  }

#pragma unroll
  for (int t = 0; t < 4; ++t)
#pragma unroll
    for (int r = 0; r < 8; ++r) {
      int row = m_blk + wave * 16 + r + half * 8;
      C[(size_t)row * N + n_blk + t * 16 + col] = acc[t][r];
    }
}

// ---------------------------------------------------------------------------
// RoPE for Q: fp32 (b,s,H*HD) -> bf16 (b,h,s,hd)
// freqs_cis: (S, HD/2, 2) = (cos, sin)
// ---------------------------------------------------------------------------
__global__ __launch_bounds__(256) void rope_q_kernel(const float* __restrict__ Qf,
                                                     const float* __restrict__ fc,
                                                     __bf16* __restrict__ Qb) {
  int idx = blockIdx.x * blockDim.x + threadIdx.x;  // B*S*H*(HD/2) = 2^22
  int i = idx & 31;
  int h = (idx >> 5) & 31;
  int s = (idx >> 10) & 2047;
  int b = idx >> 21;
  const float* src = Qf + ((size_t)(b * S_ + s)) * (H_ * HD_) + h * HD_ + 2 * i;
  float tr = src[0], ti = src[1];
  float c  = fc[(s * (HD_ / 2) + i) * 2 + 0];
  float sn = fc[(s * (HD_ / 2) + i) * 2 + 1];
  __bf16* dst = Qb + (((size_t)(b * H_ + h)) * S_ + s) * HD_ + 2 * i;
  dst[0] = (__bf16)(tr * c - ti * sn);
  dst[1] = (__bf16)(tr * sn + ti * c);
}

// RoPE for K: fp32 (b,s,KVH*HD) -> bf16 (b,kv,s,hd)
__global__ __launch_bounds__(256) void rope_k_kernel(const float* __restrict__ Kf,
                                                     const float* __restrict__ fc,
                                                     __bf16* __restrict__ Kb) {
  int idx = blockIdx.x * blockDim.x + threadIdx.x;  // B*S*KVH*(HD/2) = 2^20
  int i  = idx & 31;
  int kv = (idx >> 5) & 7;
  int s  = (idx >> 8) & 2047;
  int b  = idx >> 19;
  const float* src = Kf + ((size_t)(b * S_ + s)) * (KVH_ * HD_) + kv * HD_ + 2 * i;
  float tr = src[0], ti = src[1];
  float c  = fc[(s * (HD_ / 2) + i) * 2 + 0];
  float sn = fc[(s * (HD_ / 2) + i) * 2 + 1];
  __bf16* dst = Kb + (((size_t)(b * KVH_ + kv)) * S_ + s) * HD_ + 2 * i;
  dst[0] = (__bf16)(tr * c - ti * sn);
  dst[1] = (__bf16)(tr * sn + ti * c);
}

// V: fp32 (b,s,KVH*HD) -> bf16 transposed (b,kv,hd,s) so PV B-fragments are
// contiguous (16 consecutive key positions at fixed hd).
__global__ __launch_bounds__(256) void v_transpose_kernel(const float* __restrict__ Vf,
                                                          __bf16* __restrict__ VT) {
  int idx = blockIdx.x * blockDim.x + threadIdx.x;  // B*S*KVH*HD = 2^21
  int hd = idx & 63;
  int kv = (idx >> 6) & 7;
  int s  = (idx >> 9) & 2047;
  int b  = idx >> 20;
  VT[(((size_t)(b * KVH_ + kv)) * HD_ + hd) * S_ + s] =
      (__bf16)Vf[((size_t)(b * S_ + s)) * (KVH_ * HD_) + kv * HD_ + hd];
}

// ---------------------------------------------------------------------------
// Flash-style causal GQA attention.
// Block = 8 waves; each wave owns (b, h, 16-query block).
// Per key block of 32: scores = Q(16x64) @ K^T via 2 chained WMMAs per 16-col
// tile, causal mask, online softmax (shfl_xor row reductions inside each
// 16-lane half), P through per-wave LDS tile to get A-fragment layout, then
// 4 WMMA accumulations into O(16x64).
// ---------------------------------------------------------------------------
__global__ __launch_bounds__(256) void attn_kernel(const __bf16* __restrict__ Qb,
                                                   const __bf16* __restrict__ Kb,
                                                   const __bf16* __restrict__ VT,
                                                   __bf16* __restrict__ attn) {
  __shared__ __align__(16) __bf16 plds[8 * 16 * 32];  // per-wave 16x32 P tile

  const int lane = threadIdx.x & 31;
  const int wave = threadIdx.x >> 5;
  const int idx  = blockIdx.x * 8 + wave;  // 0 .. B*H*(S/16)-1 = 8191
  const int qt = idx & 127;                // S/16 = 128
  const int h  = (idx >> 7) & 31;
  const int b  = idx >> 12;
  const int kv = h >> 2;                   // G = 4
  const int q0 = qt * 16;
  const int n    = lane & 15;
  const int half = lane >> 4;
  __bf16* pw = plds + wave * 512;

  // Q A-fragments (hd 0..31 and 32..63), lane row = q0 + n
  const __bf16* qrow =
      Qb + (((size_t)(b * H_ + h)) * S_ + q0 + n) * HD_;
  v16bf qf0 = load_frag(qrow + half * 16,      qrow + half * 16 + 8);
  v16bf qf1 = load_frag(qrow + 32 + half * 16, qrow + 32 + half * 16 + 8);

  const __bf16* Kbase = Kb + ((size_t)(b * KVH_ + kv)) * S_ * HD_;
  const __bf16* Vbase = VT + ((size_t)(b * KVH_ + kv)) * HD_ * S_;

  float m[8], l[8];
  v8f o[4] = {};
#pragma unroll
  for (int r = 0; r < 8; ++r) { m[r] = -__builtin_inff(); l[r] = 0.0f; }

  const float scale = 0.125f;  // 1/sqrt(64)
  const int nkb = (q0 + 15) / 32 + 1;

  for (int kb = 0; kb < nkb; ++kb) {
    const int jb = kb * 32;

    // ---- scores: two 16x16 tiles (keys jb+n and jb+16+n) ----
    const __bf16* krow0 = Kbase + (size_t)(jb + n) * HD_;
    const __bf16* krow1 = Kbase + (size_t)(jb + 16 + n) * HD_;
    v16bf k00 = load_frag(krow0 + half * 16,      krow0 + half * 16 + 8);
    v16bf k01 = load_frag(krow0 + 32 + half * 16, krow0 + 32 + half * 16 + 8);
    v16bf k10 = load_frag(krow1 + half * 16,      krow1 + half * 16 + 8);
    v16bf k11 = load_frag(krow1 + 32 + half * 16, krow1 + 32 + half * 16 + 8);

    v8f s0 = {};
    s0 = wmma_bf16(qf0, k00, s0);
    s0 = wmma_bf16(qf1, k01, s0);
    v8f s1 = {};
    s1 = wmma_bf16(qf0, k10, s1);
    s1 = wmma_bf16(qf1, k11, s1);

    // ---- mask + online softmax (row r lives in one 16-lane half) ----
#pragma unroll
    for (int r = 0; r < 8; ++r) {
      const int qrow_i = q0 + r + half * 8;
      float v0 = s0[r] * scale;
      float v1 = s1[r] * scale;
      if (jb + n > qrow_i)      v0 = -__builtin_inff();
      if (jb + 16 + n > qrow_i) v1 = -__builtin_inff();

      float rmax = fmaxf(v0, v1);
      rmax = fmaxf(rmax, __shfl_xor(rmax, 1, 32));
      rmax = fmaxf(rmax, __shfl_xor(rmax, 2, 32));
      rmax = fmaxf(rmax, __shfl_xor(rmax, 4, 32));
      rmax = fmaxf(rmax, __shfl_xor(rmax, 8, 32));

      const float mnew  = fmaxf(m[r], rmax);
      const float alpha = __expf(m[r] - mnew);
      const float p0 = __expf(v0 - mnew);
      const float p1 = __expf(v1 - mnew);
      float rs = p0 + p1;
      rs += __shfl_xor(rs, 1, 32);
      rs += __shfl_xor(rs, 2, 32);
      rs += __shfl_xor(rs, 4, 32);
      rs += __shfl_xor(rs, 8, 32);

      l[r] = l[r] * alpha + rs;
      m[r] = mnew;
      o[0][r] = o[0][r] * alpha;
      o[1][r] = o[1][r] * alpha;
      o[2][r] = o[2][r] * alpha;
      o[3][r] = o[3][r] * alpha;

      // stash P tile into LDS (C layout -> row-major 16x32)
      pw[(r + half * 8) * 32 + n]      = (__bf16)p0;
      pw[(r + half * 8) * 32 + 16 + n] = (__bf16)p1;
    }

    // ---- P (16x32) @ V (32x64): re-read P as A-fragment from LDS ----
    // Same-wave DS ops are in-order, so no barrier is needed.
    const __bf16* prow = pw + n * 32;
    v16bf pf = load_frag(prow + half * 16, prow + half * 16 + 8);

#pragma unroll
    for (int t = 0; t < 4; ++t) {
      const __bf16* vp = Vbase + (size_t)(t * 16 + n) * S_ + jb + half * 16;
      v16bf vf = load_frag(vp, vp + 8);
      o[t] = wmma_bf16(pf, vf, o[t]);
    }
  }

  // ---- normalize and write attn output (b, s, H*HD) in bf16 ----
#pragma unroll
  for (int t = 0; t < 4; ++t)
#pragma unroll
    for (int r = 0; r < 8; ++r) {
      const int q = q0 + r + half * 8;
      attn[((size_t)(b * S_ + q)) * (H_ * HD_) + h * HD_ + t * 16 + n] =
          (__bf16)(o[t][r] / l[r]);
    }
}

// ---------------------------------------------------------------------------
// Host launch
// ---------------------------------------------------------------------------
extern "C" void kernel_launch(void* const* d_in, const int* in_sizes, int n_in,
                              void* d_out, int out_size, void* d_ws, size_t ws_size,
                              hipStream_t stream) {
  (void)in_sizes; (void)n_in; (void)out_size; (void)ws_size;
  const float* x  = (const float*)d_in[0];
  const float* fc = (const float*)d_in[1];
  // d_in[2] = causal mask (implicit in attention kernel)
  const float* Wq = (const float*)d_in[3];
  const float* Wk = (const float*)d_in[4];
  const float* Wv = (const float*)d_in[5];
  const float* Wo = (const float*)d_in[6];
  float* out = (float*)d_out;

  char* ws = (char*)d_ws;
  auto alloc = [&](size_t bytes) -> char* {
    char* p = ws;
    ws += (bytes + 255) & ~(size_t)255;
    return p;
  };

  const int M = B_ * S_;       // 4096 rows
  const int NQ = H_ * HD_;     // 2048
  const int NK = KVH_ * HD_;   // 512

  __bf16* xb   = (__bf16*)alloc((size_t)M * D_ * 2);
  __bf16* WqT  = (__bf16*)alloc((size_t)NQ * D_ * 2);
  __bf16* WkT  = (__bf16*)alloc((size_t)NK * D_ * 2);
  __bf16* WvT  = (__bf16*)alloc((size_t)NK * D_ * 2);
  __bf16* WoT  = (__bf16*)alloc((size_t)D_ * NQ * 2);
  float*  Qf   = (float*)alloc((size_t)M * NQ * 4);
  float*  Kf   = (float*)alloc((size_t)M * NK * 4);
  float*  Vf   = (float*)alloc((size_t)M * NK * 4);
  __bf16* Qb   = (__bf16*)alloc((size_t)M * NQ * 2);
  __bf16* Kb   = (__bf16*)alloc((size_t)M * NK * 2);
  __bf16* VT   = (__bf16*)alloc((size_t)M * NK * 2);
  __bf16* attn = (__bf16*)alloc((size_t)M * NQ * 2);

  const int TPB = 256;

  // 1) precision / layout prep
  cvt_f32_bf16<<<(M * D_) / TPB, TPB, 0, stream>>>(x, xb, M * D_);
  transpose_bf16<<<(D_ * NQ) / TPB, TPB, 0, stream>>>(Wq, WqT, D_, NQ);
  transpose_bf16<<<(D_ * NK) / TPB, TPB, 0, stream>>>(Wk, WkT, D_, NK);
  transpose_bf16<<<(D_ * NK) / TPB, TPB, 0, stream>>>(Wv, WvT, D_, NK);
  transpose_bf16<<<(NQ * D_) / TPB, TPB, 0, stream>>>(Wo, WoT, NQ, D_);

  // 2) QKV projections (LDS-staged WMMA GEMMs)
  {
    dim3 gq(NQ / BN, M / BM);
    gemm_bf16_wmma<<<gq, TPB, 0, stream>>>(xb, WqT, Qf, M, NQ, D_);
    dim3 gk(NK / BN, M / BM);
    gemm_bf16_wmma<<<gk, TPB, 0, stream>>>(xb, WkT, Kf, M, NK, D_);
    gemm_bf16_wmma<<<gk, TPB, 0, stream>>>(xb, WvT, Vf, M, NK, D_);
  }

  // 3) RoPE + attention-friendly layouts
  rope_q_kernel<<<(B_ * S_ * H_ * (HD_ / 2)) / TPB, TPB, 0, stream>>>(Qf, fc, Qb);
  rope_k_kernel<<<(B_ * S_ * KVH_ * (HD_ / 2)) / TPB, TPB, 0, stream>>>(Kf, fc, Kb);
  v_transpose_kernel<<<(B_ * S_ * KVH_ * HD_) / TPB, TPB, 0, stream>>>(Vf, VT);

  // 4) flash attention (WMMA scores + WMMA PV)
  attn_kernel<<<(B_ * H_ * (S_ / 16)) / 8, TPB, 0, stream>>>(Qb, Kb, VT, attn);

  // 5) output projection (WMMA GEMM) -> fp32 result
  {
    dim3 go(D_ / BN, M / BM);
    gemm_bf16_wmma<<<go, TPB, 0, stream>>>(attn, WoT, out, M, D_, NQ);
  }
}
